// E3Conv_30408368455708
// MI455X (gfx1250) — matrix operand order repaired and checked by reading, not compiled, not run on hardware
//
#include <hip/hip_runtime.h>
#include <cmath>

typedef _Float16 v16h __attribute__((ext_vector_type(16)));
typedef float    v8f  __attribute__((ext_vector_type(8)));

#define NPATH 17
#define W3J_TOTAL 1077

// (l1,l2,l3) per path, flat offsets of each dense (2l1+1)x(2l2+1)x(2l3+1) block
constexpr int PL1[NPATH]  = {0,0,0,1,1,1,1,1,2,2,2,2,2,3,3,3,3};
constexpr int PL2[NPATH]  = {0,1,2,0,1,1,2,2,0,1,1,2,2,0,1,2,2};
constexpr int PL3[NPATH]  = {0,1,2,1,0,2,1,3,2,1,3,0,2,3,2,1,3};
constexpr int POFF[NPATH] = {0,1,10,35,44,53,98,143,248,273,318,423,448,573,622,727,832};
constexpr int SL0[4]      = {0,1,4,9};

// PATH_W = sqrt((2*l3+1)/K_OUT[l3]),  K_OUT = {0:3,1:5,2:5,3:4}
__constant__ float c_PATH_W[NPATH] = {
    0.57735027f, 0.77459667f, 1.0f, 0.77459667f, 0.57735027f, 1.0f, 0.77459667f,
    1.32287566f, 1.0f, 0.77459667f, 1.32287566f, 0.57735027f, 1.0f, 1.32287566f,
    1.0f, 0.77459667f, 1.32287566f};

// --------------------------- W3J precompute (FP64, tiny) ---------------------------
__device__ __forceinline__ double dfact(int n) {
    const double f[11] = {1., 1., 2., 6., 24., 120., 720., 5040., 40320., 362880., 3628800.};
    return f[n];
}

__device__ double cg_coeff(int j1, int m1, int j2, int m2, int j3, int m3) {
    if (m1 + m2 != m3) return 0.0;
    double pre = sqrt((double)(2 * j3 + 1) * dfact(j3 + j1 - j2) * dfact(j3 - j1 + j2) *
                      dfact(j1 + j2 - j3) / dfact(j1 + j2 + j3 + 1));
    pre *= sqrt(dfact(j3 + m3) * dfact(j3 - m3) * dfact(j1 - m1) * dfact(j1 + m1) *
                dfact(j2 - m2) * dfact(j2 + m2));
    int k0 = 0;
    if (j2 - j3 - m1 > k0) k0 = j2 - j3 - m1;
    if (j1 - j3 + m2 > k0) k0 = j1 - j3 + m2;
    int k1 = j1 + j2 - j3;
    if (j1 - m1 < k1) k1 = j1 - m1;
    if (j2 + m2 < k1) k1 = j2 + m2;
    double s = 0.0;
    for (int k = k0; k <= k1; ++k) {
        double t = 1.0 / (dfact(k) * dfact(j1 + j2 - j3 - k) * dfact(j1 - m1 - k) *
                          dfact(j2 + m2 - k) * dfact(j3 - j2 + m1 + k) * dfact(j3 - j1 - m2 + k));
        s += (k & 1) ? -t : t;
    }
    return pre * s;
}

// entry (row a, col i) of complex change-of-basis Q_l (includes (-1j)^l factor)
__device__ void qmat(int l, int a, int i, double* qre, double* qim) {
    const double s2 = 0.70710678118654752440;
    int m = a - l;
    double qr = 0.0, qi = 0.0;
    if (m < 0) {
        if (i == l - m) qr = s2;   // col l+|m|
        if (i == l + m) qi = -s2;  // col l-|m|
    } else if (m == 0) {
        if (i == l) qr = 1.0;
    } else {
        double sg = (m & 1) ? -1.0 : 1.0;
        if (i == l + m) qr = sg * s2;
        if (i == l - m) qi = sg * s2;
    }
    double fr, fi;
    switch (l & 3) {  // (-i)^l
        case 0: fr = 1;  fi = 0;  break;
        case 1: fr = 0;  fi = -1; break;
        case 2: fr = -1; fi = 0;  break;
        default: fr = 0; fi = 1;  break;
    }
    *qre = fr * qr - fi * qi;
    *qim = fr * qi + fi * qr;
}

__global__ void __launch_bounds__(256) w3j_kernel(float* __restrict__ w3j) {
    const int p = blockIdx.x;
    const int tid = threadIdx.x;
    const int l1 = PL1[p], l2 = PL2[p], l3 = PL3[p];
    const int d1 = 2 * l1 + 1, d2 = 2 * l2 + 1, d3 = 2 * l3 + 1;
    const int cnt = d1 * d2 * d3;

    double ra_re = 0, ra_im = 0, rb_re = 0, rb_im = 0;
    if (tid < cnt) {
        const int c = tid % d3;
        const int b = (tid / d3) % d2;
        const int a = tid / (d3 * d2);
        for (int i = 0; i < d1; ++i) {
            double q1r, q1i;
            qmat(l1, a, i, &q1r, &q1i);
            if (q1r == 0.0 && q1i == 0.0) continue;
            for (int j = 0; j < d2; ++j) {
                double q2r, q2i;
                qmat(l2, b, j, &q2r, &q2i);
                if (q2r == 0.0 && q2i == 0.0) continue;
                double tr = q1r * q2r - q1i * q2i;
                double ti = q1r * q2i + q1i * q2r;
                for (int k = 0; k < d3; ++k) {
                    double cgv = cg_coeff(l1, i - l1, l2, j - l2, l3, k - l3);
                    if (cgv == 0.0) continue;
                    double q3r, q3i;
                    qmat(l3, c, k, &q3r, &q3i);
                    ra_re += (tr * q3r + ti * q3i) * cgv;  // * conj(Q3)
                    ra_im += (ti * q3r - tr * q3i) * cgv;
                    rb_re += (tr * q3r - ti * q3i) * cgv;  // * Q3
                    rb_im += (tr * q3i + ti * q3r) * cgv;
                }
            }
        }
    }

    __shared__ double red[256];
    red[tid] = fabs(ra_im);
    __syncthreads();
    for (int s = 128; s > 0; s >>= 1) { if (tid < s) red[tid] += red[tid + s]; __syncthreads(); }
    double sumA = red[0];
    __syncthreads();
    red[tid] = fabs(rb_im);
    __syncthreads();
    for (int s = 128; s > 0; s >>= 1) { if (tid < s) red[tid] += red[tid + s]; __syncthreads(); }
    double sumB = red[0];
    __syncthreads();
    double rr = (sumA <= sumB) ? ra_re : rb_re;
    red[tid] = rr * rr;
    __syncthreads();
    for (int s = 128; s > 0; s >>= 1) { if (tid < s) red[tid] += red[tid + s]; __syncthreads(); }
    double nrm = sqrt(red[0]);
    if (tid < cnt) w3j[POFF[p] + tid] = (float)(rr / nrm);
}

__global__ void zero_kernel(float* __restrict__ p, int n) {
    int i = blockIdx.x * 256 + threadIdx.x;
    if (i < n) p[i] = 0.0f;
}

// --------------------------- Main fused edge kernel ---------------------------
// 256 threads = 8 waves; each wave owns 32 edges = two 16-row WMMA M-tiles.
__global__ void __launch_bounds__(256) e3conv_kernel(
    const float* __restrict__ f_in, const float* __restrict__ pos,
    const int* __restrict__ esrc, const int* __restrict__ edst,
    const float* __restrict__ W1, const float* __restrict__ W2,
    const float* __restrict__ w3j, float* __restrict__ out,
    int E, float inv_nn) {
    __shared__ _Float16 s_W1h[8][64];       // layer-1 weights (f16)
    __shared__ _Float16 s_W2h[64][17];      // layer-2 weights * PATH_W/8 (f16)
    __shared__ _Float16 s_emb[8][32][8];    // per-wave radial embeddings
    __shared__ _Float16 s_h[8][32][64];     // per-wave hidden activations
    __shared__ float    s_w[8][32][18];     // per-wave 17 path weights (padded)
    __shared__ float    s_w3j[W3J_TOTAL];   // CG tensors (broadcast reads)

    const int tid = threadIdx.x;
    const int wave = tid >> 5;
    const int lane = tid & 31;
    const int lc = lane & 15;   // column / row-in-tile id
    const int lhw = lane >> 4;  // half-wave id

    // ---- stage weights + W3J into LDS ----
    for (int i = tid; i < 8 * 64; i += 256) s_W1h[i >> 6][i & 63] = (_Float16)W1[i];
    for (int i = tid; i < 64 * 17; i += 256) {
        int k = i / 17, p = i % 17;
        s_W2h[k][p] = (_Float16)(W2[i] * c_PATH_W[p] * 0.125f);  // fold 1/sqrt(64) & PATH_W
    }
    for (int i = tid; i < W3J_TOTAL; i += 256) s_w3j[i] = w3j[i];

    // ---- phase 1: per-edge geometry, SH(l<=2), Gaussian radial embedding ----
    const int e = blockIdx.x * 256 + wave * 32 + lane;
    const bool valid = (e < E);
    const int ec = valid ? e : (E - 1);
    const int src = esrc[ec];
    const int dst = edst[ec];
    float dx = pos[3 * dst + 0] - pos[3 * src + 0];
    float dy = pos[3 * dst + 1] - pos[3 * src + 1];
    float dz = pos[3 * dst + 2] - pos[3 * src + 2];
    float r2 = dx * dx + dy * dy + dz * dz;
    float ir = __builtin_amdgcn_rsqf(r2);  // 1/r  (single v_rsq_f32)
    float r = r2 * ir;                     // r = r2/sqrt(r2)
    float ux = dx * ir, uy = dy * ir, uz = dz * ir;
    const float s3 = 1.7320508075688772f, s5 = 2.23606797749979f;
    float sh[9];
    sh[0] = 1.0f;
    sh[1] = s3 * uy;
    sh[2] = s3 * uz;
    sh[3] = s3 * ux;
    sh[4] = s5 * s3 * ux * uy;
    sh[5] = s5 * s3 * uy * uz;
    sh[6] = s5 * 0.5f * (3.0f * uz * uz - 1.0f);
    sh[7] = s5 * s3 * ux * uz;
    sh[8] = s5 * 0.5f * s3 * (ux * ux - uy * uy);
    const float stepc = 6.0f / 9.0f;
    const float inv_step = 1.5f;  // exact 1/step
#pragma unroll
    for (int k = 0; k < 8; ++k) {
        // fold sqrt(8)/1.12 (emb) * 1/sqrt(8) (layer-1) = 1/1.12
        float g = (r - (float)(k + 1) * stepc) * inv_step;
        s_emb[wave][lane][k] = (_Float16)(__expf(-g * g) * (1.0f / 1.12f));
    }
    __syncthreads();

    // ---- phase 2: layer-1 GEMM via WMMA (16x16x32 f16, K=8 zero-padded) + SiLU ----
#pragma unroll
    for (int t = 0; t < 2; ++t) {
        v16h a;
#pragma unroll
        for (int j = 0; j < 16; ++j) a[j] = (_Float16)0.0f;
        if (lhw == 0) {  // lanes 0-15 carry K=0..7; lanes 16-31 carry K=8..15 (all zero)
#pragma unroll
            for (int j = 0; j < 8; ++j) a[j] = s_emb[wave][t * 16 + lc][j];
        }
#pragma unroll
        for (int nb = 0; nb < 4; ++nb) {
            v16h b;
#pragma unroll
            for (int j = 0; j < 16; ++j) b[j] = (_Float16)0.0f;
            if (lhw == 0) {  // B: K=0..15 on lanes 0-15 (only K<8 nonzero)
#pragma unroll
                for (int j = 0; j < 8; ++j) b[j] = s_W1h[j][nb * 16 + lc];
            }
            v8f c;
#pragma unroll
            for (int j = 0; j < 8; ++j) c[j] = 0.0f;
            c = __builtin_amdgcn_wmma_f32_16x16x32_f16(false, a, false, b, (short)0, c, false, false);
            const int nCol = nb * 16 + lc;
            const int mB = lhw * 8;
#pragma unroll
            for (int rr = 0; rr < 8; ++rr) {
                float x = c[rr];
                // SILU_MOM * x * sigmoid(x), sigmoid via fast rcp (v_rcp_f32)
                float sig = __builtin_amdgcn_rcpf(1.0f + __expf(-x));
                float hv = 1.679f * x * sig;
                s_h[wave][t * 16 + mB + rr][nCol] = (_Float16)hv;
            }
        }
    }
    __syncthreads();

    // ---- phase 3: layer-2 GEMM via WMMA (K=64 as 2x32, N=17 padded to 32) ----
    {
        const int hiK = lhw * 8;
#pragma unroll
        for (int t = 0; t < 2; ++t) {
            v16h a0, a1;
#pragma unroll
            for (int j = 0; j < 8; ++j) {
                a0[j]     = s_h[wave][t * 16 + lc][hiK + j];
                a0[j + 8] = s_h[wave][t * 16 + lc][hiK + 16 + j];
                a1[j]     = s_h[wave][t * 16 + lc][32 + hiK + j];
                a1[j + 8] = s_h[wave][t * 16 + lc][32 + hiK + 16 + j];
            }
#pragma unroll
            for (int nb = 0; nb < 2; ++nb) {
                const int col = nb * 16 + lc;
                const int kb = lhw * 16;
                v16h b0, b1;
#pragma unroll
                for (int j = 0; j < 16; ++j) { b0[j] = (_Float16)0.0f; b1[j] = (_Float16)0.0f; }
                if (col < 17) {
#pragma unroll
                    for (int j = 0; j < 16; ++j) {
                        b0[j] = s_W2h[hiK + (j & 8) + j][col];
                        b1[j] = s_W2h[32 + hiK + (j & 8) + j][col];
                    }
                }
                v8f c;
#pragma unroll
                for (int j = 0; j < 8; ++j) c[j] = 0.0f;
                c = __builtin_amdgcn_wmma_f32_16x16x32_f16(false, a0, false, b0, (short)0, c, false, false);
                c = __builtin_amdgcn_wmma_f32_16x16x32_f16(false, a1, false, b1, (short)0, c, false, false);
                if (col < 17) {
#pragma unroll
                    for (int rr = 0; rr < 8; ++rr)
                        s_w[wave][t * 16 + lhw * 8 + rr][col] = c[rr];
                }
            }
        }
    }
    __syncthreads();

    // ---- phase 4: per-edge Clebsch-Gordan tensor product (fully unrolled) ----
    float wv[NPATH];
#pragma unroll
    for (int p = 0; p < NPATH; ++p) wv[p] = s_w[wave][lane][p];
    float fv[16];
    const float* fp = f_in + (size_t)src * 16;
#pragma unroll
    for (int i = 0; i < 16; ++i) fv[i] = fp[i];
    float acc[16];
#pragma unroll
    for (int i = 0; i < 16; ++i) acc[i] = 0.0f;

#pragma unroll
    for (int p = 0; p < NPATH; ++p) {
        const int l1 = PL1[p], l2 = PL2[p], l3 = PL3[p];
        const int d1 = 2 * l1 + 1, d2 = 2 * l2 + 1, d3 = 2 * l3 + 1;
        const int o1 = SL0[l1], o2 = SL0[l2], o3 = SL0[l3];
        const int base = POFF[p];
#pragma unroll
        for (int a = 0; a < d1; ++a) {
#pragma unroll
            for (int b = 0; b < d2; ++b) {
                const float fb = fv[o1 + a] * sh[o2 + b] * wv[p];
#pragma unroll
                for (int cc = 0; cc < d3; ++cc) {
                    acc[o3 + cc] += fb * s_w3j[base + (a * d2 + b) * d3 + cc];
                }
            }
        }
    }

    // ---- phase 5: segment-sum via global f32 atomics ----
    if (valid) {
        float* op = out + (size_t)dst * 16;
#pragma unroll
        for (int i = 0; i < 16; ++i) atomicAdd(op + i, acc[i] * inv_nn);
    }
}

extern "C" void kernel_launch(void* const* d_in, const int* in_sizes, int n_in,
                              void* d_out, int out_size, void* d_ws, size_t ws_size,
                              hipStream_t stream) {
    const float* f_in = (const float*)d_in[0];
    const float* pos  = (const float*)d_in[1];
    const int*   esrc = (const int*)d_in[2];
    const int*   edst = (const int*)d_in[3];
    const float* W1   = (const float*)d_in[4];
    const float* W2   = (const float*)d_in[5];
    float* out = (float*)d_out;
    float* w3j = (float*)d_ws;

    const int E = in_sizes[2];
    const int N = in_sizes[1] / 3;
    const float inv_nn = (float)sqrt((double)N / (double)E);  // 1/sqrt(E/N)

    zero_kernel<<<(out_size + 255) / 256, 256, 0, stream>>>(out, out_size);
    w3j_kernel<<<NPATH, 256, 0, stream>>>(w3j);
    const int blocks = (E + 255) / 256;
    e3conv_kernel<<<blocks, 256, 0, stream>>>(f_in, pos, esrc, edst, W1, W2, w3j, out, E, inv_nn);
}
